// InnerAttention_54528904790349
// MI455X (gfx1250) — matrix operand order repaired and checked by reading, not compiled, not run on hardware
//
#include <hip/hip_runtime.h>
#include <hip/hip_bf16.h>

typedef __attribute__((ext_vector_type(16))) __bf16 v16bf;
typedef __attribute__((ext_vector_type(8)))  float  v8f;
typedef __attribute__((ext_vector_type(4)))  int    v4i;

constexpr int kB  = 2;
constexpr int kH  = 8;
constexpr int kD  = 64;
constexpr int kN  = 2048;
constexpr int kC  = 512;
constexpr int kF3 = 3 * kH * kD;      // 1536
constexpr int kM  = kB * kN;          // 4096
constexpr float kScale = 0.125f;      // 64^-0.5
constexpr float kLog2e = 1.4426950408889634f;

#if defined(__has_builtin)
#if __has_builtin(__builtin_amdgcn_global_load_async_to_lds_b128)
#define HAVE_ASYNC_LDS 1
#endif
#if __has_builtin(__builtin_amdgcn_s_wait_asynccnt)
#define HAVE_WAIT_ASYNC 1
#endif
#endif

union FragBF {
  v16bf v;
  uint4 u[2];
};

__device__ __forceinline__ unsigned short f2bf(float f) {
  unsigned int u = __float_as_uint(f);
  u += 0x7FFFu + ((u >> 16) & 1u);     // round-to-nearest-even
  return (unsigned short)(u >> 16);
}

__device__ __forceinline__ void wait_async_zero() {
#if defined(HAVE_ASYNC_LDS)
#if defined(HAVE_WAIT_ASYNC)
  __builtin_amdgcn_s_wait_asynccnt(0);
#else
  asm volatile("s_wait_asynccnt 0x0" ::: "memory");
#endif
#endif
}

// ---------------------------------------------------------------------------
// 1) f32 -> bf16 conversion
// ---------------------------------------------------------------------------
__global__ void f2bf_kernel(const float* __restrict__ src,
                            unsigned short* __restrict__ dst, int n) {
  int i = blockIdx.x * blockDim.x + threadIdx.x;
  if (i < n) dst[i] = f2bf(src[i]);
}

// ---------------------------------------------------------------------------
// Register-blocked wave GEMM: 32x64 output tile per wave.
// C[i][j] (i<2 M-subtiles, j<4 F-subtiles) = A[m0+32, :K] * B[f0+64, :K]^T.
// A,B bf16 row-major, leading dim = K. Per 32-deep K step: 4 A-loads +
// 8 B-loads (b128) feed 8 v_wmma_f32_16x16x32_bf16 -> 1.5 loads/WMMA.
// A-frag: lane<16 holds K{0..7,16..23} of row lane%16; B-frag: lane%16 = out
// column, lane/16 selects K{0..7,16..23}/{8..15,24..31}, same as A.
// ---------------------------------------------------------------------------
__device__ __forceinline__ void wave_gemm_32x64(const unsigned short* __restrict__ A,
                                                int lda, int m0,
                                                const unsigned short* __restrict__ Bw,
                                                int ldb, int f0,
                                                int K, int lane,
                                                v8f acc[2][4]) {
  const int klo  = (lane < 16) ? 0 : 8;
  const int half = lane >> 4;
  const unsigned short* arow0 = A + (long)(m0 + (lane & 15)) * lda;
  const unsigned short* arow1 = arow0 + (long)16 * lda;
  const unsigned short* brow0 = Bw + (long)(f0 + (lane & 15)) * ldb;

  for (int k0 = 0; k0 < K; k0 += 32) {
    FragBF a0, a1;
    a0.u[0] = *(const uint4*)(arow0 + k0 + klo);
    a0.u[1] = *(const uint4*)(arow0 + k0 + 16 + klo);
    a1.u[0] = *(const uint4*)(arow1 + k0 + klo);
    a1.u[1] = *(const uint4*)(arow1 + k0 + 16 + klo);
#pragma unroll
    for (int j = 0; j < 4; ++j) {
      FragBF b;
      const unsigned short* bp = brow0 + (long)j * 16 * ldb + k0 + klo;
      b.u[0] = *(const uint4*)(bp);
      b.u[1] = *(const uint4*)(bp + 16);
      acc[0][j] = __builtin_amdgcn_wmma_f32_16x16x32_bf16(
          false, a0.v, false, b.v, (short)0, acc[0][j], false, false);
      acc[1][j] = __builtin_amdgcn_wmma_f32_16x16x32_bf16(
          false, a1.v, false, b.v, (short)0, acc[1][j], false, false);
    }
  }
}

// ---------------------------------------------------------------------------
// 2) QKV projection: qkv[m, f] = x[m,:] . w_qkv[f,:] + b_qkv[f]
//    f = (t*H + h)*D + d ; writes Qs (f32, pre-scaled), K/V (bf16) in [B,H,N,D]
//    Each wave: 32 rows x 64 features (one whole (t,h) d-block).
// ---------------------------------------------------------------------------
__global__ void qkv_gemm_kernel(const unsigned short* __restrict__ xb,
                                const unsigned short* __restrict__ wb,
                                const float* __restrict__ bias,
                                float* __restrict__ Qs,
                                unsigned short* __restrict__ Kb,
                                unsigned short* __restrict__ Vb) {
  const int lane = threadIdx.x & 31;
  const int wid  = threadIdx.x >> 5;
  const int tileF = blockIdx.x * 8 + wid;   // 0..23  (64-wide feature blocks)
  const int tileM = blockIdx.y;             // 0..127 (32-row blocks)
  const int f0 = tileF * 64, m0 = tileM * 32;

  v8f acc[2][4];
#pragma unroll
  for (int i = 0; i < 2; ++i)
#pragma unroll
    for (int j = 0; j < 4; ++j) { v8f z = {}; acc[i][j] = z; }

  wave_gemm_32x64(xb, kC, m0, wb, kC, f0, kC, lane, acc);

  const int t = f0 / (kH * kD);             // feature block is one (t,h)
  const int h = (f0 % (kH * kD)) / kD;
  const int roff = (lane >> 4) << 3;
#pragma unroll
  for (int j = 0; j < 4; ++j) {
    const float bv = bias[f0 + j * 16 + (lane & 15)];
    const int d = j * 16 + (lane & 15);
#pragma unroll
    for (int i = 0; i < 2; ++i) {
#pragma unroll
      for (int r = 0; r < 8; ++r) {
        int m = m0 + i * 16 + roff + r;
        int b = m >> 11, n = m & (kN - 1);
        float v = acc[i][j][r] + bv;
        long idx = (((long)(b * kH + h) * kN) + n) * kD + d;
        if (t == 0)      Qs[idx] = v * kScale;
        else if (t == 1) Kb[idx] = f2bf(v);
        else             Vb[idx] = f2bf(v);
      }
    }
  }
}

// ---------------------------------------------------------------------------
// 3) EPEG folded into Q: Q'[n] = Qs[n] + sum_j conv_w[h,j] * Qs[n+j-2]
//    (conv on the score rows commutes with the K contraction; conv_b is a
//     per-row constant and cancels in softmax). Output bf16.
// ---------------------------------------------------------------------------
__global__ void convq_kernel(const float* __restrict__ Qs,
                             const float* __restrict__ cw,
                             unsigned short* __restrict__ Qp) {
  int i = blockIdx.x * blockDim.x + threadIdx.x;  // over B*H*N*D
  int d  = i & (kD - 1);
  int n  = (i >> 6) & (kN - 1);
  int bh = i >> 17;                               // / (N*D)
  int h  = bh & (kH - 1);
  float acc = Qs[i];
#pragma unroll
  for (int j = 0; j < 5; ++j) {
    int nn = n + j - 2;
    if (nn >= 0 && nn < kN)
      acc += cw[h * 5 + j] * Qs[((long)(bh << 11) + nn) * kD + d];
  }
  Qp[i] = f2bf(acc);
}

// ---------------------------------------------------------------------------
// 4) Fused flash attention. Block = 8 waves = 128 queries per (b,h).
//    K tile (64 keys x 64 d) pulled into LDS with GLOBAL_LOAD_ASYNC_TO_LDS
//    (ASYNCcnt) when available; V staged transposed ([d][key]) so PV
//    B-fragments are contiguous b128 LDS reads. Online softmax in f32,
//    P re-laid-out C-frag -> A-frag through per-wave LDS staging.
// ---------------------------------------------------------------------------
__global__ void flash_attn_kernel(const unsigned short* __restrict__ Qp,
                                  const unsigned short* __restrict__ Kb,
                                  const unsigned short* __restrict__ Vb,
                                  unsigned short* __restrict__ AO) {
  __shared__ unsigned short Ks[64 * 72];      // [key][d], padded stride
  __shared__ unsigned short Vt[64 * 72];      // [d][key], padded stride
  __shared__ unsigned short Ps[8][16 * 72];   // per-wave P staging

  const int lane = threadIdx.x & 31;
  const int wid  = threadIdx.x >> 5;
  const int bh   = blockIdx.y;
  const int b    = bh >> 3, h = bh & 7;
  const int qbase = blockIdx.x * 128 + wid * 16;

  const int klo  = (lane < 16) ? 0 : 8;
  const int half = lane >> 4;

  // Preload A-fragments of Q' for this wave's 16 rows (K-chunks d=0..31,32..63)
  FragBF aQ[2];
  {
    const unsigned short* qrow =
        Qp + ((long)bh * kN + qbase + (lane & 15)) * kD;
#pragma unroll
    for (int c = 0; c < 2; ++c) {
      aQ[c].u[0] = *(const uint4*)(qrow + c * 32 + half * 16);
      aQ[c].u[1] = *(const uint4*)(qrow + c * 32 + half * 16 + 8);
    }
  }

  v8f O[4];
  float mrun[8], lrun[8];
#pragma unroll
  for (int t = 0; t < 4; ++t) { v8f z = {}; O[t] = z; }
#pragma unroll
  for (int r = 0; r < 8; ++r) { mrun[r] = -1e30f; lrun[r] = 0.0f; }

  const unsigned short* Kbh = Kb + (long)bh * kN * kD;
  const unsigned short* Vbh = Vb + (long)bh * kN * kD;

  const int ldrow = threadIdx.x >> 2;          // 0..63
  const int ldcol = (threadIdx.x & 3) * 16;    // 0,16,32,48

  for (int k0 = 0; k0 < kN; k0 += 64) {
    // ---- K tile: async copy to LDS when available --------------------------
    {
      const unsigned short* gk = Kbh + (long)(k0 + ldrow) * kD + ldcol;
#if defined(HAVE_ASYNC_LDS)
      __builtin_amdgcn_global_load_async_to_lds_b128(
          (v4i*)gk, (v4i*)&Ks[ldrow * 72 + ldcol], 0, 0);
      __builtin_amdgcn_global_load_async_to_lds_b128(
          (v4i*)(gk + 8), (v4i*)&Ks[ldrow * 72 + ldcol + 8], 0, 0);
#else
      uint4 k0v = *(const uint4*)(gk);
      uint4 k1v = *(const uint4*)(gk + 8);
      *(uint4*)(&Ks[ldrow * 72 + ldcol])     = k0v;
      *(uint4*)(&Ks[ldrow * 72 + ldcol + 8]) = k1v;
#endif
      // ---- V tile: load + transpose-store ([d][key]) -----------------------
      uint4 v0v = *(const uint4*)(Vbh + (long)(k0 + ldrow) * kD + ldcol);
      uint4 v1v = *(const uint4*)(Vbh + (long)(k0 + ldrow) * kD + ldcol + 8);
      const unsigned short* p0 = (const unsigned short*)&v0v;
      const unsigned short* p1 = (const unsigned short*)&v1v;
#pragma unroll
      for (int e = 0; e < 8; ++e) {
        Vt[(ldcol + e) * 72 + ldrow]     = p0[e];
        Vt[(ldcol + 8 + e) * 72 + ldrow] = p1[e];
      }
    }
    wait_async_zero();
    __syncthreads();

    // ---- S = Q' . K^T  over 4 key sub-tiles of 16 --------------------------
    v8f S[4];
#pragma unroll
    for (int t = 0; t < 4; ++t) {
      const unsigned short* kp = &Ks[(t * 16 + (lane & 15)) * 72 + half * 16];
      FragBF bK0, bK1;
      bK0.u[0] = *(const uint4*)(kp + 0);
      bK0.u[1] = *(const uint4*)(kp + 8);
      bK1.u[0] = *(const uint4*)(kp + 32);
      bK1.u[1] = *(const uint4*)(kp + 40);
      v8f s = {};
      s = __builtin_amdgcn_wmma_f32_16x16x32_bf16(false, aQ[0].v, false, bK0.v,
                                                  (short)0, s, false, false);
      s = __builtin_amdgcn_wmma_f32_16x16x32_bf16(false, aQ[1].v, false, bK1.v,
                                                  (short)0, s, false, false);
      S[t] = s;
    }

    // ---- Online softmax (row = r + 8*(lane/16); 16 lanes share a row) ------
    float pr[4][8];
#pragma unroll
    for (int r = 0; r < 8; ++r) {
      float mx = fmaxf(fmaxf(S[0][r], S[1][r]), fmaxf(S[2][r], S[3][r]));
#pragma unroll
      for (int msk = 8; msk >= 1; msk >>= 1)
        mx = fmaxf(mx, __shfl_xor(mx, msk, 32));
      float mnew = fmaxf(mrun[r], mx);
      float corr = exp2f((mrun[r] - mnew) * kLog2e);
      float rs = 0.0f;
#pragma unroll
      for (int t = 0; t < 4; ++t) {
        float p = exp2f((S[t][r] - mnew) * kLog2e);
        pr[t][r] = p;
        rs += p;
      }
#pragma unroll
      for (int msk = 8; msk >= 1; msk >>= 1)
        rs += __shfl_xor(rs, msk, 32);
      lrun[r] = lrun[r] * corr + rs;
      mrun[r] = mnew;
#pragma unroll
      for (int t = 0; t < 4; ++t) O[t][r] *= corr;
    }

    // ---- P: C-layout f32 -> A-layout bf16 via per-wave LDS staging ---------
    unsigned short* myP = &Ps[wid][0];
    {
      int roff = (lane >> 4) << 3;
#pragma unroll
      for (int t = 0; t < 4; ++t)
#pragma unroll
        for (int r = 0; r < 8; ++r)
          myP[(roff + r) * 72 + t * 16 + (lane & 15)] = f2bf(pr[t][r]);
    }
    FragBF aP[2];
    {
      const unsigned short* prow = &myP[(lane & 15) * 72];
#pragma unroll
      for (int c = 0; c < 2; ++c) {
        aP[c].u[0] = *(const uint4*)(prow + c * 32 + half * 16);
        aP[c].u[1] = *(const uint4*)(prow + c * 32 + half * 16 + 8);
      }
    }

    // ---- O += P . V  (4 output-d sub-tiles, contraction over 64 keys) ------
#pragma unroll
    for (int t = 0; t < 4; ++t) {
      const unsigned short* vp = &Vt[(t * 16 + (lane & 15)) * 72 + half * 16];
      FragBF bV0, bV1;
      bV0.u[0] = *(const uint4*)(vp + 0);
      bV0.u[1] = *(const uint4*)(vp + 8);
      bV1.u[0] = *(const uint4*)(vp + 32);
      bV1.u[1] = *(const uint4*)(vp + 40);
      O[t] = __builtin_amdgcn_wmma_f32_16x16x32_bf16(false, aP[0].v, false, bV0.v,
                                                     (short)0, O[t], false, false);
      O[t] = __builtin_amdgcn_wmma_f32_16x16x32_bf16(false, aP[1].v, false, bV1.v,
                                                     (short)0, O[t], false, false);
    }
    __syncthreads();
  }

  // ---- Normalize, write O as bf16 in [B, N, H*D] ---------------------------
  {
    int roff = (lane >> 4) << 3;
#pragma unroll
    for (int r = 0; r < 8; ++r) {
      float inv = 1.0f / lrun[r];
#pragma unroll
      for (int t = 0; t < 4; ++t) {
        int n = qbase + roff + r;
        int col = h * kD + t * 16 + (lane & 15);
        AO[((long)b * kN + n) * (kH * kD) + col] = f2bf(O[t][r] * inv);
      }
    }
  }
}

// ---------------------------------------------------------------------------
// 5) Output projection: out[m, c] = AO[m,:] . w_proj[c,:] + b_proj[c]  (f32)
//    Each wave: 32 rows x 64 output channels.
// ---------------------------------------------------------------------------
__global__ void proj_gemm_kernel(const unsigned short* __restrict__ AO,
                                 const unsigned short* __restrict__ wb,
                                 const float* __restrict__ bias,
                                 float* __restrict__ out) {
  const int lane = threadIdx.x & 31;
  const int wid  = threadIdx.x >> 5;
  const int tileC = wid;                    // 0..7 (64-wide channel blocks)
  const int tileM = blockIdx.y;             // 0..127
  const int c0 = tileC * 64, m0 = tileM * 32;

  v8f acc[2][4];
#pragma unroll
  for (int i = 0; i < 2; ++i)
#pragma unroll
    for (int j = 0; j < 4; ++j) { v8f z = {}; acc[i][j] = z; }

  wave_gemm_32x64(AO, kC, m0, wb, kC, c0, kC, lane, acc);

  const int roff = (lane >> 4) << 3;
#pragma unroll
  for (int j = 0; j < 4; ++j) {
    const float bv = bias[c0 + j * 16 + (lane & 15)];
#pragma unroll
    for (int i = 0; i < 2; ++i)
#pragma unroll
      for (int r = 0; r < 8; ++r)
        out[(long)(m0 + i * 16 + roff + r) * kC + c0 + j * 16 + (lane & 15)] =
            acc[i][j][r] + bv;
  }
}

// ---------------------------------------------------------------------------
extern "C" void kernel_launch(void* const* d_in, const int* in_sizes, int n_in,
                              void* d_out, int out_size, void* d_ws, size_t ws_size,
                              hipStream_t stream) {
  const float* x      = (const float*)d_in[0];
  const float* w_qkv  = (const float*)d_in[1];
  const float* b_qkv  = (const float*)d_in[2];
  const float* w_proj = (const float*)d_in[3];
  const float* b_proj = (const float*)d_in[4];
  const float* conv_w = (const float*)d_in[5];
  // d_in[6] (conv_b): per-head constant added to every logit of a row ->
  // cancels in softmax; unused.

  char* ws = (char*)d_ws;
  unsigned short* xb     = (unsigned short*)(ws + 0);          //  4.0 MB
  unsigned short* wqkvb  = (unsigned short*)(ws + 4194304);    //  1.5 MB
  unsigned short* wprojb = (unsigned short*)(ws + 5767168);    //  0.5 MB
  float*          Qs     = (float*)         (ws + 6291456);    //  8.0 MB
  unsigned short* Qp     = (unsigned short*)(ws + 14680064);   //  4.0 MB
  unsigned short* Kb     = (unsigned short*)(ws + 18874368);   //  4.0 MB
  unsigned short* Vb     = (unsigned short*)(ws + 23068672);   //  4.0 MB
  unsigned short* AO     = (unsigned short*)(ws + 27262976);   //  4.0 MB

  f2bf_kernel<<<(kM * kC + 255) / 256, 256, 0, stream>>>(x, xb, kM * kC);
  f2bf_kernel<<<(kF3 * kC + 255) / 256, 256, 0, stream>>>(w_qkv, wqkvb, kF3 * kC);
  f2bf_kernel<<<(kC * kC + 255) / 256, 256, 0, stream>>>(w_proj, wprojb, kC * kC);

  // 24 feature-blocks of 64 x 128 row-blocks of 32, 8 waves/block
  qkv_gemm_kernel<<<dim3(3, 128), 256, 0, stream>>>(xb, wqkvb, b_qkv, Qs, Kb, Vb);

  convq_kernel<<<(kB * kH * kN * kD) / 256, 256, 0, stream>>>(Qs, conv_w, Qp);

  flash_attn_kernel<<<dim3(kN / 128, kB * kH), 256, 0, stream>>>(Qp, Kb, Vb, AO);

  // 8 channel-blocks of 64 (one per wave) x 128 row-blocks of 32
  proj_gemm_kernel<<<dim3(1, 128), 256, 0, stream>>>(AO, wprojb, b_proj,
                                                     (float*)d_out);
}
